// Decoder_3040836846277
// MI455X (gfx1250) — matrix operand order repaired and checked by reading, not compiled, not run on hardware
//
#include <hip/hip_runtime.h>
#include <hip/hip_bf16.h>
#include <stdint.h>

// ---------------------------------------------------------------------------
// Shapes (fixed by the reference)
// ---------------------------------------------------------------------------
#define LB 256      // batch
#define LT 21       // timesteps (1 feature step + 20 caption steps)
#define LL 20       // caption length
#define LE 512      // embed dim
#define LH 1024     // hidden
#define LV 10000    // vocab
#define BN_EPS 1e-5f

typedef __attribute__((ext_vector_type(16))) __bf16 v16bf;
typedef __attribute__((ext_vector_type(8)))  float  v8f;

// ---------------------------------------------------------------------------
// WMMA fragment loaders (bf16, 16x16x32, wave32)
//
// A (16x32, M x K), dense 16-bit layout (ISA 7.12.2):
//   all lanes: row = row0 + (lane & 15)
//   VGPR v (v<4):  K = 8*half + 2v, +1      (half = lane>>4)
//   VGPR v (v>=4): K = 16 + 8*half + 2(v-4), +1
// => per lane: two 16B chunks at k0 + 8*half and k0 + 8*half + 16.
//
// B (32x16, K x N): lane&15 = N column; same K order as A => two 16B
// chunks per lane of W[n][k0 + 8*half ..] and W[n][k0 + 8*half + 16 ..]
// (W row-major [N][K], B[k][n] = W[n][k]).
// ---------------------------------------------------------------------------
__device__ __forceinline__ v16bf ld_a_frag(const __bf16* rowp, int k0, int half) {
  const uint32_t* p0 = (const uint32_t*)(rowp + k0 + 8 * half);
  const uint32_t* p1 = (const uint32_t*)(rowp + k0 + 8 * half + 16);
  union { v16bf v; uint32_t u[8]; } f;
#pragma unroll
  for (int i = 0; i < 4; ++i) { f.u[i] = p0[i]; f.u[4 + i] = p1[i]; }
  return f.v;
}

__device__ __forceinline__ v16bf ld_b_frag(const __bf16* rowp, int k0, int half) {
  const uint32_t* q0 = (const uint32_t*)(rowp + k0 + 8 * half);
  const uint32_t* q1 = (const uint32_t*)(rowp + k0 + 8 * half + 16);
  union { v16bf v; uint32_t u[8]; } f;
#pragma unroll
  for (int i = 0; i < 4; ++i) { f.u[i] = q0[i]; f.u[4 + i] = q1[i]; }
  return f.v;
}

#define WMMA_BF16(a, b, c) \
  __builtin_amdgcn_wmma_f32_16x16x32_bf16(false, (a), false, (b), (short)0, (c), false, false)

// ---------------------------------------------------------------------------
// One K-segment of a 32(M) x 64(N) wave tile.
// Ping-pong double buffering via unroll-by-64: two statically distinct
// fragment buffer sets, each refilled IN PLACE on alternate half-iterations.
// No buffer rotation moves -> no VGPR blowup; loads for one buffer are in
// flight while the 8 WMMAs of the other buffer execute.
// Requires K % 64 == 0 (holds: K in {512, 1024}).
// ---------------------------------------------------------------------------
__device__ __forceinline__ void gemm_seg_32x64(
    const __bf16* __restrict__ a0row, const __bf16* __restrict__ a1row,
    const __bf16* const brow[4], int K, int half, v8f acc[4][2]) {
  // Buffer set 0: preload k = 0
  v16bf A0_0 = ld_a_frag(a0row, 0, half);
  v16bf A1_0 = ld_a_frag(a1row, 0, half);
  v16bf B0_0 = ld_b_frag(brow[0], 0, half);
  v16bf B1_0 = ld_b_frag(brow[1], 0, half);
  v16bf B2_0 = ld_b_frag(brow[2], 0, half);
  v16bf B3_0 = ld_b_frag(brow[3], 0, half);

  for (int k0 = 0; k0 < K; k0 += 64) {
    // Buffer set 1: load k0+32 (always valid: K % 64 == 0)
    const int kb = k0 + 32;
    v16bf A0_1 = ld_a_frag(a0row, kb, half);
    v16bf A1_1 = ld_a_frag(a1row, kb, half);
    v16bf B0_1 = ld_b_frag(brow[0], kb, half);
    v16bf B1_1 = ld_b_frag(brow[1], kb, half);
    v16bf B2_1 = ld_b_frag(brow[2], kb, half);
    v16bf B3_1 = ld_b_frag(brow[3], kb, half);

    // Compute on buffer set 0 (k0)
    acc[0][0] = WMMA_BF16(A0_0, B0_0, acc[0][0]);
    acc[0][1] = WMMA_BF16(A1_0, B0_0, acc[0][1]);
    acc[1][0] = WMMA_BF16(A0_0, B1_0, acc[1][0]);
    acc[1][1] = WMMA_BF16(A1_0, B1_0, acc[1][1]);
    acc[2][0] = WMMA_BF16(A0_0, B2_0, acc[2][0]);
    acc[2][1] = WMMA_BF16(A1_0, B2_0, acc[2][1]);
    acc[3][0] = WMMA_BF16(A0_0, B3_0, acc[3][0]);
    acc[3][1] = WMMA_BF16(A1_0, B3_0, acc[3][1]);

    // Refill buffer set 0 with k0+64 (skip on last iteration)
    const int kc = k0 + 64;
    if (kc < K) {
      A0_0 = ld_a_frag(a0row, kc, half);
      A1_0 = ld_a_frag(a1row, kc, half);
      B0_0 = ld_b_frag(brow[0], kc, half);
      B1_0 = ld_b_frag(brow[1], kc, half);
      B2_0 = ld_b_frag(brow[2], kc, half);
      B3_0 = ld_b_frag(brow[3], kc, half);
    }

    // Compute on buffer set 1 (k0+32)
    acc[0][0] = WMMA_BF16(A0_1, B0_1, acc[0][0]);
    acc[0][1] = WMMA_BF16(A1_1, B0_1, acc[0][1]);
    acc[1][0] = WMMA_BF16(A0_1, B1_1, acc[1][0]);
    acc[1][1] = WMMA_BF16(A1_1, B1_1, acc[1][1]);
    acc[2][0] = WMMA_BF16(A0_1, B2_1, acc[2][0]);
    acc[2][1] = WMMA_BF16(A1_1, B2_1, acc[2][1]);
    acc[3][0] = WMMA_BF16(A0_1, B3_1, acc[3][0]);
    acc[3][1] = WMMA_BF16(A1_1, B3_1, acc[3][1]);
  }
}

// ---------------------------------------------------------------------------
// f32 -> bf16 weight conversion
// ---------------------------------------------------------------------------
__global__ void cvt_bf16_kernel(const float* __restrict__ src,
                                __bf16* __restrict__ dst, size_t n) {
  size_t i = (size_t)blockIdx.x * blockDim.x + threadIdx.x;
  if (i < n) dst[i] = (__bf16)src[i];
}

// ---------------------------------------------------------------------------
// Build xs[T][B][E] (bf16, time-major): t=0 -> features, t>=1 -> embedding
// ---------------------------------------------------------------------------
__global__ void embed_concat_kernel(const float* __restrict__ features,
                                    const int* __restrict__ captions,
                                    const float* __restrict__ table,
                                    __bf16* __restrict__ xs) {
  int tb = blockIdx.x;          // [0, LT*LB)
  int t = tb / LB, b = tb % LB;
  const float* src;
  if (t == 0) {
    src = features + (size_t)b * LE;
  } else {
    int tok = captions[b * LL + (t - 1)];
    src = table + (size_t)tok * LE;
  }
  __bf16* dst = xs + ((size_t)t * LB + b) * LE;
  for (int e = threadIdx.x; e < LE; e += blockDim.x) dst[e] = (__bf16)src[e];
}

// ---------------------------------------------------------------------------
// gates = A1 @ B1^T + A2 @ B2^T + (bias_a + bias_b)   [M x N], f32 out
// One wave computes a 32x64 tile: 8 WMMAs per K-step, B fragments reused 2x.
// ---------------------------------------------------------------------------
__global__ void __launch_bounds__(256, 1)
gates_gemm_kernel(const __bf16* __restrict__ A1,
                  const __bf16* __restrict__ B1, int K1,
                  const __bf16* __restrict__ A2,
                  const __bf16* __restrict__ B2, int K2,
                  const float* __restrict__ bias_a,
                  const float* __restrict__ bias_b,
                  float* __restrict__ C, int M, int N) {
  const int lane = threadIdx.x & 31;
  const int half = lane >> 4;
  const int ln   = lane & 15;
  const int w = blockIdx.x * (blockDim.x >> 5) + (threadIdx.x >> 5);
  const int mtiles = M >> 5;                  // 32-row tiles
  const int ntiles = N >> 6;                  // 64-col tiles
  if (w >= mtiles * ntiles) return;           // wave-uniform
  const int mt = w % mtiles, nt = w / mtiles;
  const int row0 = mt << 5, col0 = nt << 6;

  v8f acc[4][2];
#pragma unroll
  for (int nf = 0; nf < 4; ++nf) {
    int c = col0 + 16 * nf + ln;
    float bv = bias_a[c] + bias_b[c];
#pragma unroll
    for (int j = 0; j < 8; ++j) { acc[nf][0][j] = bv; acc[nf][1][j] = bv; }
  }

  {
    const __bf16* brow[4];
#pragma unroll
    for (int nf = 0; nf < 4; ++nf)
      brow[nf] = B1 + (size_t)(col0 + 16 * nf + ln) * K1;
    gemm_seg_32x64(A1 + (size_t)(row0 + ln) * K1,
                   A1 + (size_t)(row0 + 16 + ln) * K1, brow, K1, half, acc);
  }
  {
    const __bf16* brow[4];
#pragma unroll
    for (int nf = 0; nf < 4; ++nf)
      brow[nf] = B2 + (size_t)(col0 + 16 * nf + ln) * K2;
    gemm_seg_32x64(A2 + (size_t)(row0 + ln) * K2,
                   A2 + (size_t)(row0 + 16 + ln) * K2, brow, K2, half, acc);
  }

  // C/D layout: VGPR j, lanes 0-15 -> M=j, lanes 16-31 -> M=j+8; N = lane&15
#pragma unroll
  for (int nf = 0; nf < 4; ++nf) {
    int c = col0 + 16 * nf + ln;
#pragma unroll
    for (int j = 0; j < 8; ++j) {
      C[(size_t)(row0 + j + 8 * half) * N + c]      = acc[nf][0][j];
      C[(size_t)(row0 + 16 + j + 8 * half) * N + c] = acc[nf][1][j];
    }
  }
}

// ---------------------------------------------------------------------------
// LSTM cell elementwise: gates [B,4H] (i,f,g,o) -> h (f32 + bf16), c (f32)
// ---------------------------------------------------------------------------
__device__ __forceinline__ float sigf(float x) { return 1.f / (1.f + __expf(-x)); }

__global__ void lstm_cell_kernel(const float* __restrict__ gates,
                                 float* __restrict__ c,
                                 float* __restrict__ hf,
                                 __bf16* __restrict__ hb) {
  int idx = blockIdx.x * blockDim.x + threadIdx.x;
  if (idx >= LB * LH) return;
  int b = idx / LH, j = idx % LH;
  const float* g = gates + (size_t)b * 4 * LH;
  float ig = sigf(g[j]);
  float fg = sigf(g[j + LH]);
  float gg = tanhf(g[j + 2 * LH]);
  float og = sigf(g[j + 3 * LH]);
  float cn = fg * c[idx] + ig * gg;
  c[idx] = cn;
  float hn = og * tanhf(cn);
  hf[idx] = hn;
  hb[idx] = (__bf16)hn;
}

// ---------------------------------------------------------------------------
// BatchNorm (train, per-step batch stats, biased var) + locked-dropout mask.
// One block per column (H blocks, B=256 threads). Output bf16.
// ---------------------------------------------------------------------------
__global__ void bn_mask_kernel(const float* __restrict__ h,
                               const float* __restrict__ gamma,
                               const float* __restrict__ beta,
                               const float* __restrict__ mask,
                               __bf16* __restrict__ y) {
  const int col = blockIdx.x;
  const int b   = threadIdx.x;          // 0..255
  float x = h[(size_t)b * LH + col];
  __shared__ float ssum[LB];
  __shared__ float ssq[LB];
  ssum[b] = x;
  ssq[b]  = x * x;
  __syncthreads();
  for (int s = LB >> 1; s > 0; s >>= 1) {
    if (b < s) { ssum[b] += ssum[b + s]; ssq[b] += ssq[b + s]; }
    __syncthreads();
  }
  float mu  = ssum[0] * (1.f / LB);
  float var = ssq[0] * (1.f / LB) - mu * mu;
  float yn  = gamma[col] * (x - mu) * rsqrtf(var + BN_EPS) + beta[col];
  y[(size_t)b * LH + col] = (__bf16)(yn * mask[(size_t)b * LH + col]);
}

// ---------------------------------------------------------------------------
// Output projection: out[m, n] = sum_k Y[m, k] * Wout[n, k] + b_out[n]
// Y time-major [T][B][H]; packed output row m = b*T + t. 32x64 wave tiles.
// OOB N-fragments are column-CLAMPED to N-16 so loads stay legal and the
// inner loop is branch-free; only bias-init and stores are guarded
// (guards are fragment-uniform since N % 16 == 0 -> EXEC all-ones at WMMA).
// ---------------------------------------------------------------------------
__global__ void __launch_bounds__(256, 1)
out_gemm_kernel(const __bf16* __restrict__ Y,
                const __bf16* __restrict__ W,
                const float* __restrict__ bias,
                float* __restrict__ C) {
  const int M = LB * LT, N = LV, K = LH;
  const int lane = threadIdx.x & 31;
  const int half = lane >> 4;
  const int ln   = lane & 15;
  const int w = blockIdx.x * (blockDim.x >> 5) + (threadIdx.x >> 5);
  const int mtiles = M >> 5;                 // 168
  const int ntiles = (N + 63) >> 6;          // 157
  if (w >= mtiles * ntiles) return;          // wave-uniform
  const int mt = w % mtiles, nt = w / mtiles;
  const int row0 = mt << 5, col0 = nt << 6;

  // A-row remap: packed row m = b*T + t  ->  Y[t][b][:]
  const int m0 = row0 + ln;
  const int m1 = row0 + 16 + ln;
  const __bf16* a0row = Y + ((size_t)(m0 % LT) * LB + (m0 / LT)) * K;
  const __bf16* a1row = Y + ((size_t)(m1 % LT) * LB + (m1 / LT)) * K;

  const __bf16* brow[4];
#pragma unroll
  for (int nf = 0; nf < 4; ++nf) {
    int cb = col0 + 16 * nf;
    if (cb > N - 16) cb = N - 16;            // clamp: loads legal, acc unused
    brow[nf] = W + (size_t)(cb + ln) * K;
  }

  v8f acc[4][2];
#pragma unroll
  for (int nf = 0; nf < 4; ++nf) {
    int cb = col0 + 16 * nf;
    float bv = (cb < N) ? bias[cb + ln] : 0.f;
#pragma unroll
    for (int j = 0; j < 8; ++j) { acc[nf][0][j] = bv; acc[nf][1][j] = bv; }
  }

  gemm_seg_32x64(a0row, a1row, brow, K, half, acc);

#pragma unroll
  for (int nf = 0; nf < 4; ++nf) {
    int cb = col0 + 16 * nf;
    if (cb < N) {                            // fragment-uniform
      int c = cb + ln;
#pragma unroll
      for (int j = 0; j < 8; ++j) {
        C[(size_t)(row0 + j + 8 * half) * N + c]      = acc[nf][0][j];
        C[(size_t)(row0 + 16 + j + 8 * half) * N + c] = acc[nf][1][j];
      }
    }
  }
}

// ---------------------------------------------------------------------------
// Host orchestration
// ---------------------------------------------------------------------------
extern "C" void kernel_launch(void* const* d_in, const int* in_sizes, int n_in,
                              void* d_out, int out_size, void* d_ws, size_t ws_size,
                              hipStream_t stream) {
  (void)in_sizes; (void)n_in; (void)out_size; (void)ws_size;

  const float* features = (const float*)d_in[0];
  const int*   captions = (const int*)  d_in[1];
  // d_in[2] = lengths (unused; all == T)
  const float* table  = (const float*)d_in[3];
  const float* W_ih1  = (const float*)d_in[4];
  const float* W_hh1  = (const float*)d_in[5];
  const float* b_ih1  = (const float*)d_in[6];
  const float* b_hh1  = (const float*)d_in[7];
  const float* gamma1 = (const float*)d_in[8];
  const float* beta1  = (const float*)d_in[9];
  const float* mask1  = (const float*)d_in[10];
  const float* W_ih2  = (const float*)d_in[11];
  const float* W_hh2  = (const float*)d_in[12];
  const float* b_ih2  = (const float*)d_in[13];
  const float* b_hh2  = (const float*)d_in[14];
  const float* gamma2 = (const float*)d_in[15];
  const float* beta2  = (const float*)d_in[16];
  const float* mask2  = (const float*)d_in[17];
  const float* W_out  = (const float*)d_in[18];
  const float* b_out  = (const float*)d_in[19];

  // Workspace carve-up (256B aligned slices)
  char* base = (char*)d_ws;
  size_t off = 0;
  auto take = [&](size_t bytes) {
    size_t o = off; off += (bytes + 255) & ~(size_t)255; return o;
  };
  __bf16* xs    = (__bf16*)(base + take((size_t)LT * LB * LE * 2));
  __bf16* wih1b = (__bf16*)(base + take((size_t)4 * LH * LE * 2));
  __bf16* whh1b = (__bf16*)(base + take((size_t)4 * LH * LH * 2));
  __bf16* wih2b = (__bf16*)(base + take((size_t)4 * LH * LH * 2));
  __bf16* whh2b = (__bf16*)(base + take((size_t)4 * LH * LH * 2));
  __bf16* woutb = (__bf16*)(base + take((size_t)LV * LH * 2));
  float*  gates1 = (float*)(base + take((size_t)LB * 4 * LH * 4));
  float*  gates2 = (float*)(base + take((size_t)LB * 4 * LH * 4));
  float*  h1f = (float*)(base + take((size_t)LB * LH * 4));
  float*  c1  = (float*)(base + take((size_t)LB * LH * 4));
  float*  h2f = (float*)(base + take((size_t)LB * LH * 4));
  float*  c2  = (float*)(base + take((size_t)LB * LH * 4));
  __bf16* h1b = (__bf16*)(base + take((size_t)LB * LH * 2));
  __bf16* h2b = (__bf16*)(base + take((size_t)LB * LH * 2));
  __bf16* y1b = (__bf16*)(base + take((size_t)LB * LH * 2));
  __bf16* ys  = (__bf16*)(base + take((size_t)LT * LB * LH * 2));

  // Zero recurrent state (graph-capturable memset nodes)
  hipMemsetAsync(h1f, 0, (size_t)LB * LH * 4, stream);
  hipMemsetAsync(c1,  0, (size_t)LB * LH * 4, stream);
  hipMemsetAsync(h2f, 0, (size_t)LB * LH * 4, stream);
  hipMemsetAsync(c2,  0, (size_t)LB * LH * 4, stream);
  hipMemsetAsync(h1b, 0, (size_t)LB * LH * 2, stream);
  hipMemsetAsync(h2b, 0, (size_t)LB * LH * 2, stream);

  // Weight conversions to bf16
  auto cvt = [&](const float* s, __bf16* d, size_t n) {
    cvt_bf16_kernel<<<(unsigned)((n + 255) / 256), 256, 0, stream>>>(s, d, n);
  };
  cvt(W_ih1, wih1b, (size_t)4 * LH * LE);
  cvt(W_hh1, whh1b, (size_t)4 * LH * LH);
  cvt(W_ih2, wih2b, (size_t)4 * LH * LH);
  cvt(W_hh2, whh2b, (size_t)4 * LH * LH);
  cvt(W_out, woutb, (size_t)LV * LH);

  // Input sequence: [features ; embed(captions)] -> xs[T][B][E] bf16
  embed_concat_kernel<<<LT * LB, 256, 0, stream>>>(features, captions, table, xs);

  // Recurrent scan: 21 sequential timesteps
  const int gatesWaves  = (LB / 32) * ((4 * LH) / 64);   // 512
  const int gatesBlocks = gatesWaves / 8;                // 64 (8 waves/block)
  const int cellBlocks  = (LB * LH) / 256;               // 1024

  for (int t = 0; t < LT; ++t) {
    const __bf16* xt = xs + (size_t)t * LB * LE;
    gates_gemm_kernel<<<gatesBlocks, 256, 0, stream>>>(
        xt, wih1b, LE, h1b, whh1b, LH, b_ih1, b_hh1, gates1, LB, 4 * LH);
    lstm_cell_kernel<<<cellBlocks, 256, 0, stream>>>(gates1, c1, h1f, h1b);
    bn_mask_kernel<<<LH, LB, 0, stream>>>(h1f, gamma1, beta1, mask1, y1b);

    gates_gemm_kernel<<<gatesBlocks, 256, 0, stream>>>(
        y1b, wih2b, LH, h2b, whh2b, LH, b_ih2, b_hh2, gates2, LB, 4 * LH);
    lstm_cell_kernel<<<cellBlocks, 256, 0, stream>>>(gates2, c2, h2f, h2b);
    bn_mask_kernel<<<LH, LB, 0, stream>>>(h2f, gamma2, beta2, mask2,
                                          ys + (size_t)t * LB * LH);
  }

  // Final projection to vocab: [5376 x 10000] f32
  const int mtiles = (LB * LT) / 32;           // 168
  const int ntiles = (LV + 63) / 64;           // 157
  const int outWaves  = mtiles * ntiles;       // 26376
  const int outBlocks = (outWaves + 7) / 8;    // 3297
  out_gemm_kernel<<<outBlocks, 256, 0, stream>>>(ys, woutb, b_out, (float*)d_out);
}
